// Attention_4080218931831
// MI455X (gfx1250) — matrix operand order repaired and checked by reading, not compiled, not run on hardware
//
#include <hip/hip_runtime.h>

// Problem constants (match reference)
#define IN_FEAT 10
#define HID     32
#define BATCH   256
#define NSEQ    512

typedef __attribute__((ext_vector_type(16))) _Float16 v16h;
typedef __attribute__((ext_vector_type(8)))  _Float16 v8h;
typedef __attribute__((ext_vector_type(8)))  float    v8f;
typedef __attribute__((ext_vector_type(4)))  float    v4f;

// ---------------------------------------------------------------------------
// Kernel 1: q = s @ Qw, k = s @ Kw, stored as f16 for the WMMA stage.
// One thread per (row, m) output pair; a wave covers one row's 32 outputs so
// the 10 s-values broadcast from cache and the weight loads are coalesced.
// ---------------------------------------------------------------------------
__global__ void qk_project_kernel(const float* __restrict__ s,
                                  const float* __restrict__ Qw,
                                  const float* __restrict__ Kw,
                                  _Float16* __restrict__ qbuf,
                                  _Float16* __restrict__ kbuf) {
    int gid = blockIdx.x * blockDim.x + threadIdx.x;   // BATCH*NSEQ*HID threads
    int row = gid >> 5;                                // global row in [0, B*N)
    int m   = gid & 31;                                // hidden index
    const float* srow = s + row * IN_FEAT;
    float qa = 0.f, ka = 0.f;
#pragma unroll
    for (int kk = 0; kk < IN_FEAT; ++kk) {
        float sv = srow[kk];
        qa += sv * Qw[kk * HID + m];
        ka += sv * Kw[kk * HID + m];
    }
    qbuf[row * HID + m] = (_Float16)qa;
    kbuf[row * HID + m] = (_Float16)ka;
}

// ---------------------------------------------------------------------------
// Kernel 2: per batch, scores = q @ k^T via v_wmma_f32_16x16x32_f16,
// att = scores^2 * G, row-normalized. One wave (32 threads) owns a 16-row
// slab across all 512 columns; raw att lives in LDS until the row sums are
// known, so Gmat is read exactly once and att written exactly once.
// ---------------------------------------------------------------------------
#define LDS_STRIDE 516   // 512 + 4 pad: (8*516) % 64 == 32 -> conflict-free halves

__global__ __launch_bounds__(32)
void attention_norm_kernel(const _Float16* __restrict__ qbuf,
                           const _Float16* __restrict__ kbuf,
                           const float* __restrict__ Gmat,
                           float* __restrict__ out) {
    __shared__ float att_s[16 * LDS_STRIDE];   // 16 rows x 512 cols (+pad)
    __shared__ float inv_s[16];                // per-row 1/(sum+eps)

    const int lane = threadIdx.x & 31;
    const int half = lane >> 4;      // 0: rows M..M+7, 1: rows M+8..M+15 of C
    const int n    = lane & 15;      // column-within-tile / A-row index

    const int b    = blockIdx.y;
    const int row0 = blockIdx.x * 16;            // row slab within batch

    // --- A fragment (16x32 f16): lane holds row (row0+n),
    //     K = 16*half..16*half+15 (contiguous, matching B). ---
    const _Float16* qrow = qbuf + ((size_t)(b * NSEQ) + row0 + n) * HID;
    v8h alo = *(const v8h*)(qrow + 16 * half);
    v8h ahi = *(const v8h*)(qrow + 16 * half + 8);
    v16h afrag;
#pragma unroll
    for (int i = 0; i < 8; ++i) { afrag[i] = alo[i]; afrag[8 + i] = ahi[i]; }

    const _Float16* kbase = kbuf + (size_t)(b * NSEQ) * HID;
    const float*    Gp    = Gmat + ((size_t)b * NSEQ + row0) * NSEQ;

    float rowsum[8];
#pragma unroll
    for (int r = 0; r < 8; ++r) rowsum[r] = 0.f;

    for (int jt = 0; jt < NSEQ / 16; ++jt) {
        // B fragment (32x16 = k^T tile): lane holds column j = jt*16+n,
        // K = 16*half..16*half+15 -> 16 contiguous halfs of k row j.
        const _Float16* krow = kbase + (jt * 16 + n) * HID + 16 * half;
        v16h bfrag = *(const v16h*)krow;

        v8f c = {};
        c = __builtin_amdgcn_wmma_f32_16x16x32_f16(
                /*neg_a=*/false, afrag, /*neg_b=*/false, bfrag,
                /*c_mod=*/(short)0, c, /*reuse_a=*/false, /*reuse_b=*/false);

        const int col = jt * 16 + n;
#pragma unroll
        for (int r = 0; r < 8; ++r) {
            const int lrow = r + 8 * half;              // C layout: M = r + 8*half
            float g  = Gp[(size_t)lrow * NSEQ + col];
            float v  = c[r];
            float av = v * v * g;                       // scores^2 * G
            att_s[lrow * LDS_STRIDE + col] = av;
            rowsum[r] += av;
        }
    }

    // Reduce row sums across the 16 lanes of each half-group (wave32 shuffles).
#pragma unroll
    for (int r = 0; r < 8; ++r) {
        float v = rowsum[r];
        v += __shfl_xor(v, 1, 32);
        v += __shfl_xor(v, 2, 32);
        v += __shfl_xor(v, 4, 32);
        v += __shfl_xor(v, 8, 32);
        rowsum[r] = v;
    }
#pragma unroll
    for (int r = 0; r < 8; ++r) {
        if (n == r) inv_s[r + 8 * half] = 1.0f / (rowsum[r] + 0.001f);
    }
    __syncthreads();   // single-wave WG: lowers to waitcnt; orders LDS visibility

    // Normalized write-out: float4 per thread, fully coalesced.
    float* outp = out + ((size_t)b * NSEQ + row0) * NSEQ;
#pragma unroll 4
    for (int it = 0; it < (16 * NSEQ / 4) / 32; ++it) {   // 64 iterations
        int f4  = it * 32 + threadIdx.x;
        int row = f4 >> 7;               // 128 float4 per 512-wide row
        int col = (f4 & 127) * 4;        // LDS_STRIDE%4==0 -> 16B aligned
        v4f v   = *(const v4f*)(att_s + row * LDS_STRIDE + col);
        float inv = inv_s[row];
        v4f o = { v.x * inv, v.y * inv, v.z * inv, v.w * inv };
        *(v4f*)(outp + (size_t)row * NSEQ + col) = o;
    }
}

// ---------------------------------------------------------------------------
// Launch: inputs in setup_inputs() order: s, Gmat, Qweight, Kweight.
// d_ws holds the f16 q/k buffers (2 * 256*512*32 * 2B = 16.8 MB).
// ---------------------------------------------------------------------------
extern "C" void kernel_launch(void* const* d_in, const int* in_sizes, int n_in,
                              void* d_out, int out_size, void* d_ws, size_t ws_size,
                              hipStream_t stream) {
    const float* s    = (const float*)d_in[0];
    const float* Gmat = (const float*)d_in[1];
    const float* Qw   = (const float*)d_in[2];
    const float* Kw   = (const float*)d_in[3];
    float* out = (float*)d_out;

    _Float16* qbuf = (_Float16*)d_ws;
    _Float16* kbuf = qbuf + (size_t)BATCH * NSEQ * HID;

    const int R = BATCH * NSEQ;
    qk_project_kernel<<<(R * HID) / 256, 256, 0, stream>>>(s, Qw, Kw, qbuf, kbuf);

    dim3 grid(NSEQ / 16, BATCH);   // 32 x 256 = 8192 waves
    attention_norm_kernel<<<grid, 32, 0, stream>>>(qbuf, kbuf, Gmat, out);
}
